// CreditRiskGAT_64192581206381
// MI455X (gfx1250) — matrix operand order repaired and checked by reading, not compiled, not run on hardware
//
#include <hip/hip_runtime.h>
#include <hip/hip_bf16.h>

// ---------------------------------------------------------------------------
// CreditRiskGAT on MI455X (gfx1250, wave32).
//  * GEMMs: v_wmma_f32_16x16x32_f16, 128x128x32 block tile, 8 WMMA/wave/step.
//  * All GEMM operands pre-converted to f16 with Kp padded to mult-of-32 and
//    weights pre-transposed -> staging is pure linear 16B-aligned tiles.
//  * Tile staging via the Tensor Data Mover (tensor_load_to_lds + TENSORcnt),
//    DOUBLE-BUFFERED: next tile's DMA is issued before waiting on the current
//    pair (s_wait_tensorcnt 2), overlapping DMA with WMMA compute.
//  * Attention: wave-per-(edge,head) atomics, fused "sum(w*h)/sum(w)".
// ---------------------------------------------------------------------------

typedef _Float16 v16h __attribute__((ext_vector_type(16)));
typedef _Float16 v8h  __attribute__((ext_vector_type(8)));
typedef float    v8f  __attribute__((ext_vector_type(8)));
typedef unsigned int u32x4 __attribute__((ext_vector_type(4)));
typedef int          i32x4 __attribute__((ext_vector_type(4)));
typedef int          i32x8 __attribute__((ext_vector_type(8)));

#define BM 128
#define BN 128
#define BK 32
#define LDA 40   // padded LDS row stride in halfs (80 B); pad = 8 halfs = 4 DW
#define LDB 40

#if __has_builtin(__builtin_amdgcn_tensor_load_to_lds) && __has_builtin(__builtin_amdgcn_s_wait_tensorcnt)
#define ATH_TDM 1
#else
#define ATH_TDM 0
#endif

#if ATH_TDM
// D# group1 for a 2D f16 tile: data_size=2B, LDS pad every 16 DW by 4 DW.
__device__ __forceinline__ i32x8 tdm_g1(int td0, int td1, int tile0, int tile1,
                                        int stride0)
{
    i32x8 g;
    g[0] = (1 << 16) |      // data_size = 1 -> 2 bytes
           (1 << 20) |      // pad_enable
           (3 << 22) |      // pad_interval: 16 DWORDs (= 32 halfs = one tile row)
           (3 << 25);       // pad_amount: 4 DWORDs (= 8 halfs = LDA-BK)
    g[1] = (td0 & 0xFFFF) << 16;                             // tensor_dim0[15:0]
    g[2] = ((td0 >> 16) & 0xFFFF) | ((td1 & 0xFFFF) << 16);  // td0[31:16], td1[15:0]
    g[3] = ((td1 >> 16) & 0xFFFF) | (tile0 << 16);           // td1[31:16], tile_dim0
    g[4] = tile1 & 0xFFFF;                                   // tile_dim1 (tile_dim2=0)
    g[5] = stride0;                                          // tensor_dim0_stride[31:0]
    g[6] = 0;
    g[7] = 0;
    return g;
}

__device__ __forceinline__ void tdm_load_tile(unsigned lds_off,
                                              const _Float16* gaddr,
                                              int td0, int td1,
                                              int tile0, int tile1, int stride0)
{
    unsigned long long ga = (unsigned long long)(size_t)gaddr;
    u32x4 g0;
    g0[0] = 1u;                                   // count=1 (valid), user mode
    g0[1] = lds_off;                              // lds_addr (bytes)
    g0[2] = (unsigned)ga;                         // global_addr[31:0]
    g0[3] = (unsigned)(ga >> 32) | (2u << 30);    // global_addr[56:32] | type=2
    i32x8 g1 = tdm_g1(td0, td1, tile0, tile1, stride0);
    i32x4 z4 = {0, 0, 0, 0};
#if defined(__clang_major__) && __clang_major__ >= 23
    i32x8 z8 = {0, 0, 0, 0, 0, 0, 0, 0};
    __builtin_amdgcn_tensor_load_to_lds(g0, g1, z4, z4, z8, 0);
#else
    __builtin_amdgcn_tensor_load_to_lds(g0, g1, z4, z4, 0);
#endif
}
#endif // ATH_TDM

// ---------------- WMMA GEMM: C[M,Nc] = A[M,Kp] * Bt[Nc,Kp]^T (f16 in, f32 out)
// Kp is a multiple of BK (padded with zeros at conversion time).
__global__ __launch_bounds__(256) void gat_gemm_wmma(
    const _Float16* __restrict__ A, const _Float16* __restrict__ Bt,
    float* __restrict__ C, int M, int Kp, int Nc)
{
    __shared__ __align__(16) _Float16 As[2][BM][LDA];
    __shared__ __align__(16) _Float16 Bs[2][BN][LDB];

    const int tid   = threadIdx.x;
    const int lane  = tid & 31;
    const int wave  = tid >> 5;       // 0..7
    const int waveM = wave >> 1;      // 0..3  -> 32 rows each
    const int waveN = wave & 1;       // 0..1  -> 64 cols each
    const int lhalf = lane >> 4;      // 0/1 (lane group)
    const int lrow  = lane & 15;

    const int m0 = blockIdx.x * BM;
    const int n0 = blockIdx.y * BN;
    const int nsteps = Kp / BK;       // Kp is a multiple of BK

    v8f acc[2][4] = {};

#if ATH_TDM
    if (wave == 0) {   // prologue: DMA first tile pair into buffer 0
        tdm_load_tile((unsigned)(size_t)&As[0][0][0], &A[(size_t)m0 * Kp],
                      Kp, M - m0, BK, BM, Kp);
        tdm_load_tile((unsigned)(size_t)&Bs[0][0][0], &Bt[(size_t)n0 * Kp],
                      Kp, Nc - n0, BK, BN, Kp);
    }
#endif

    for (int s = 0; s < nsteps; ++s) {
        const int k0  = s * BK;
        const int cur = s & 1;
#if ATH_TDM
        if (wave == 0) {
            if (s + 1 < nsteps) {
                // Issue next tile pair into the other buffer, then wait only
                // for the *previous* pair (in-order TENSORcnt) -> DMA overlaps
                // this step's WMMA work.
                tdm_load_tile((unsigned)(size_t)&As[cur ^ 1][0][0],
                              &A[(size_t)m0 * Kp + k0 + BK],
                              Kp - k0 - BK, M - m0, BK, BM, Kp);
                tdm_load_tile((unsigned)(size_t)&Bs[cur ^ 1][0][0],
                              &Bt[(size_t)n0 * Kp + k0 + BK],
                              Kp - k0 - BK, Nc - n0, BK, BN, Kp);
                __builtin_amdgcn_s_wait_tensorcnt(2);
            } else {
                __builtin_amdgcn_s_wait_tensorcnt(0);
            }
        }
#else
        // Manual staging fallback: 512 16-B chunks per tile, 2/thread/matrix.
        #pragma unroll
        for (int i = 0; i < 2; ++i) {
            int chunk = tid + i * 256;
            int r = chunk >> 2, c = (chunk & 3) * 8;
            int gr = m0 + r;  gr = gr < M  ? gr : M - 1;   // clamp; never stored
            int gn = n0 + r;  gn = gn < Nc ? gn : Nc - 1;  // clamp; never stored
            *(v8h*)&As[cur][r][c] = *(const v8h*)&A [(size_t)gr * Kp + k0 + c];
            *(v8h*)&Bs[cur][r][c] = *(const v8h*)&Bt[(size_t)gn * Kp + k0 + c];
        }
#endif
        __syncthreads();   // buffer `cur` ready for all waves

        // ---- Fragments per ISA §7.12.2 (16-bit A 16x32, B 32x16 layouts).
        v16h afrag[2], bfrag[4];
        #pragma unroll
        for (int mt = 0; mt < 2; ++mt) {
            int r = waveM * 32 + mt * 16 + lrow;
            v8h lo = *reinterpret_cast<const v8h*>(&As[cur][r][8 * lhalf]);
            v8h hi = *reinterpret_cast<const v8h*>(&As[cur][r][16 + 8 * lhalf]);
            #pragma unroll
            for (int i = 0; i < 8; ++i) { afrag[mt][i] = lo[i]; afrag[mt][8 + i] = hi[i]; }
        }
        #pragma unroll
        for (int nt = 0; nt < 4; ++nt) {
            int n = waveN * 64 + nt * 16 + lrow;
            v8h lo = *reinterpret_cast<const v8h*>(&Bs[cur][n][8 * lhalf]);
            v8h hi = *reinterpret_cast<const v8h*>(&Bs[cur][n][16 + 8 * lhalf]);
            #pragma unroll
            for (int i = 0; i < 8; ++i) { bfrag[nt][i] = lo[i]; bfrag[nt][8 + i] = hi[i]; }
        }
        #pragma unroll
        for (int mt = 0; mt < 2; ++mt)
            #pragma unroll
            for (int nt = 0; nt < 4; ++nt)
                acc[mt][nt] = __builtin_amdgcn_wmma_f32_16x16x32_f16(
                    false, afrag[mt], false, bfrag[nt],
                    (short)0, acc[mt][nt], false, false);
        __syncthreads();   // all reads of buffer `cur` done -> safe to refill
    }

    // ---- Epilogue. C/D layout: c[v] -> row = v + 8*lhalf, col = lrow.
    #pragma unroll
    for (int mt = 0; mt < 2; ++mt)
        #pragma unroll
        for (int nt = 0; nt < 4; ++nt)
            #pragma unroll
            for (int v = 0; v < 8; ++v) {
                int gr = m0 + waveM * 32 + mt * 16 + 8 * lhalf + v;
                int gc = n0 + waveN * 64 + nt * 16 + lrow;
                if (gr < M && gc < Nc) C[(size_t)gr * Nc + gc] = acc[mt][nt][v];
            }
}

// ---------------- f32 -> padded f16 conversions (done once per launch)
__global__ void convert_pad_f16(const float* __restrict__ X, _Float16* __restrict__ Xh,
                                int M, int K, int Kp)
{
    size_t i = (size_t)blockIdx.x * blockDim.x + threadIdx.x;
    if (i >= (size_t)M * Kp) return;
    int r = (int)(i / Kp), c = (int)(i % Kp);
    Xh[i] = (c < K) ? (_Float16)X[(size_t)r * K + c] : (_Float16)0.f;
}

// Wt[n][c] = W[c][n] (transpose + pad K->Kp with zeros)
__global__ void convert_wt_f16(const float* __restrict__ W, _Float16* __restrict__ Wt,
                               int K, int Kp, int Ncols)
{
    size_t i = (size_t)blockIdx.x * blockDim.x + threadIdx.x;
    if (i >= (size_t)Ncols * Kp) return;
    int n = (int)(i / Kp), c = (int)(i % Kp);
    Wt[i] = (c < K) ? (_Float16)W[(size_t)c * Ncols + n] : (_Float16)0.f;
}

// ---------------- order-preserving float<->uint encoding for atomic max
__device__ __forceinline__ unsigned int encf(float f) {
    unsigned int u = __float_as_uint(f);
    return (u & 0x80000000u) ? ~u : (u | 0x80000000u);
}
__device__ __forceinline__ float decf(unsigned int u) {
    u = (u & 0x80000000u) ? (u & 0x7FFFFFFFu) : ~u;
    return __uint_as_float(u);
}
#define NEG_INF_ENC 0x007FFFFFu   // encf(-inf)

__global__ void fill_u32(unsigned int* p, size_t n, unsigned int v) {
    size_t i = (size_t)blockIdx.x * blockDim.x + threadIdx.x;
    if (i < n) p[i] = v;
}

// es/ed[n,h] = dot(h[n,h,:], a_src/a_dst[h,:])
__global__ void node_scores(const float* __restrict__ h,
                            const float* __restrict__ a_src,
                            const float* __restrict__ a_dst,
                            float* __restrict__ es, float* __restrict__ ed,
                            int N, int H, int C)
{
    int i = blockIdx.x * blockDim.x + threadIdx.x;  // n*H + h
    if (i >= N * H) return;
    int hh = i % H;
    const float* hp = &h[(size_t)i * C];
    const float* as = &a_src[(size_t)hh * C];
    const float* ad = &a_dst[(size_t)hh * C];
    float s1 = 0.f, s2 = 0.f;
    for (int c = 0; c < C; ++c) { float v = hp[c]; s1 += v * as[c]; s2 += v * ad[c]; }
    es[i] = s1; ed[i] = s2;
}

// pass 1: segment max of leaky_relu(es[src]+ed[dst]) over dst
__global__ void edge_max(const int* __restrict__ src, const int* __restrict__ dst,
                         const float* __restrict__ es, const float* __restrict__ ed,
                         unsigned int* __restrict__ segmax,
                         int E, int H, long long total)
{
    long long i = (long long)blockIdx.x * blockDim.x + threadIdx.x;
    if (i >= total) return;
    int eidx = (int)(i / H), hh = (int)(i % H);
    int s, d;
    if (eidx < E) { s = src[eidx]; d = dst[eidx]; } else { s = d = eidx - E; }
    float e = es[(size_t)s * H + hh] + ed[(size_t)d * H + hh];
    e = e > 0.f ? e : 0.2f * e;
    atomicMax(&segmax[(size_t)d * H + hh], encf(e));
}

// pass 2 (wave per (edge,head)): w = exp(e-m); segsum += w; acc[dst] += w*h[src]
__global__ void edge_accum(const int* __restrict__ src, const int* __restrict__ dst,
                           const float* __restrict__ es, const float* __restrict__ ed,
                           const unsigned int* __restrict__ segmax,
                           float* __restrict__ segsum,
                           const float* __restrict__ h, float* __restrict__ acc,
                           int E, int H, int C, long long totalWaves)
{
    long long wid = ((long long)blockIdx.x * blockDim.x + threadIdx.x) >> 5;
    long long stride = ((long long)gridDim.x * blockDim.x) >> 5;
    int lane = threadIdx.x & 31;
    for (; wid < totalWaves; wid += stride) {
        int eidx = (int)(wid / H), hh = (int)(wid % H);
        int s, d;
        if (eidx < E) { s = src[eidx]; d = dst[eidx]; } else { s = d = eidx - E; }
        float e = es[(size_t)s * H + hh] + ed[(size_t)d * H + hh];
        e = e > 0.f ? e : 0.2f * e;
        float m = decf(segmax[(size_t)d * H + hh]);
        float w = __expf(e - m);
        if (lane == 0) atomicAdd(&segsum[(size_t)d * H + hh], w);
        const float* hs = &h[((size_t)s * H + hh) * C];
        float* ap = &acc[((size_t)d * H + hh) * C];
        for (int c = lane; c < C; c += 32) atomicAdd(&ap[c], w * hs[c]);
    }
}

// outh = (f16) elu(acc / (segsum + 1e-16) + bias)  — feeds the next GEMM
__global__ void node_finalize(const float* __restrict__ acc,
                              const float* __restrict__ segsum,
                              const float* __restrict__ bias,
                              _Float16* __restrict__ outh,
                              int N, int H, int C)
{
    size_t i = (size_t)blockIdx.x * blockDim.x + threadIdx.x;
    size_t total = (size_t)N * H * C;
    if (i >= total) return;
    int hc = (int)(i % (size_t)(H * C));
    int nh = (int)(i / C);             // n*H + h
    float z = segsum[nh];
    float v = acc[i] / (z + 1e-16f) + bias[hc];
    v = v > 0.f ? v : __expf(v) - 1.f;   // ELU(alpha=1)
    outh[i] = (_Float16)v;
}

// out[n] = sigmoid(dot(h3[n,:], Wr) + br)
__global__ void risk_head(const _Float16* __restrict__ h, const float* __restrict__ Wr,
                          const float* __restrict__ br, float* __restrict__ out,
                          int N, int C)
{
    int n = blockIdx.x * blockDim.x + threadIdx.x;
    if (n >= N) return;
    const _Float16* hp = &h[(size_t)n * C];
    float s = 0.f;
    for (int c = 0; c < C; ++c) s += (float)hp[c] * Wr[c];
    s += br[0];
    out[n] = 1.f / (1.f + __expf(-s));
}

extern "C" void kernel_launch(void* const* d_in, const int* in_sizes, int n_in,
                              void* d_out, int out_size, void* d_ws, size_t ws_size,
                              hipStream_t stream)
{
    const float* x   = (const float*)d_in[0];
    const int*   ei  = (const int*)d_in[1];
    const float* W1  = (const float*)d_in[2];
    const float* as1 = (const float*)d_in[3];
    const float* ad1 = (const float*)d_in[4];
    const float* b1  = (const float*)d_in[5];
    const float* W2  = (const float*)d_in[6];
    const float* as2 = (const float*)d_in[7];
    const float* ad2 = (const float*)d_in[8];
    const float* b2  = (const float*)d_in[9];
    const float* W3  = (const float*)d_in[10];
    const float* as3 = (const float*)d_in[11];
    const float* ad3 = (const float*)d_in[12];
    const float* b3  = (const float*)d_in[13];
    const float* Wr  = (const float*)d_in[14];
    const float* br  = (const float*)d_in[15];

    const int FIN = 66, FINP = 96, HMAX = 8;
    const int N = in_sizes[0] / FIN;
    const int E = in_sizes[1] / 2;
    const int Etot = E + N;                  // add_self_loops
    const int* srcI = ei;
    const int* dstI = ei + E;

    // ---- workspace layout (256-B aligned slices)
    char* ws = (char*)d_ws;
    size_t off = 0;
    auto take = [&](size_t bytes) -> char* {
        char* p = ws + off;
        off += (bytes + 255) & ~(size_t)255;
        return p;
    };
    float*     bufA = (float*)take((size_t)N * 1024 * sizeof(float));       // h (f32)
    float*     bufB = (float*)take((size_t)N * 1024 * sizeof(float));       // acc (f32)
    _Float16*  bufH = (_Float16*)take((size_t)N * 1024 * sizeof(_Float16)); // f16 acts
    _Float16*  w1t  = (_Float16*)take((size_t)1024 * FINP * sizeof(_Float16));
    _Float16*  w2t  = (_Float16*)take((size_t)128 * 1024 * sizeof(_Float16));
    _Float16*  w3t  = (_Float16*)take((size_t)64 * 128 * sizeof(_Float16));
    float* esb  = (float*)take((size_t)N * HMAX * sizeof(float));
    float* edb  = (float*)take((size_t)N * HMAX * sizeof(float));
    unsigned int* smax = (unsigned int*)take((size_t)N * HMAX * sizeof(unsigned int));
    float* ssum = (float*)take((size_t)N * HMAX * sizeof(float));
    (void)ws_size; (void)n_in;

    // ---- one-time f16 conversions (weights transposed, K padded)
    {
        size_t nx = (size_t)N * FINP;
        convert_pad_f16<<<(int)((nx + 255) / 256), 256, 0, stream>>>(x, bufH, N, FIN, FINP);
        size_t n1 = (size_t)1024 * FINP;
        convert_wt_f16<<<(int)((n1 + 255) / 256), 256, 0, stream>>>(W1, w1t, FIN, FINP, 1024);
        size_t n2 = (size_t)128 * 1024;
        convert_wt_f16<<<(int)((n2 + 255) / 256), 256, 0, stream>>>(W2, w2t, 1024, 1024, 128);
        size_t n3 = (size_t)64 * 128;
        convert_wt_f16<<<(int)((n3 + 255) / 256), 256, 0, stream>>>(W3, w3t, 128, 128, 64);
    }

    auto run_layer = [&](const _Float16* xinH, int Kp, const _Float16* WtH,
                         const float* a_s, const float* a_d, const float* bias,
                         int H, int C, float* hbuf, float* accbuf, _Float16* outH) {
        int Nc = H * C;
        dim3 g((N + BM - 1) / BM, (Nc + BN - 1) / BN);
        gat_gemm_wmma<<<g, 256, 0, stream>>>(xinH, WtH, hbuf, N, Kp, Nc);

        int NH = N * H;
        node_scores<<<(NH + 255) / 256, 256, 0, stream>>>(hbuf, a_s, a_d, esb, edb, N, H, C);

        fill_u32<<<(NH + 255) / 256, 256, 0, stream>>>(smax, (size_t)NH, NEG_INF_ENC);
        fill_u32<<<(NH + 255) / 256, 256, 0, stream>>>((unsigned int*)ssum, (size_t)NH, 0u);
        size_t accN = (size_t)N * Nc;
        fill_u32<<<(int)((accN + 255) / 256), 256, 0, stream>>>((unsigned int*)accbuf, accN, 0u);

        long long totE = (long long)Etot * H;
        edge_max<<<(int)((totE + 255) / 256), 256, 0, stream>>>(
            srcI, dstI, esb, edb, smax, E, H, totE);
        edge_accum<<<2048, 256, 0, stream>>>(
            srcI, dstI, esb, edb, smax, ssum, hbuf, accbuf, E, H, C, totE);
        node_finalize<<<(int)((accN + 255) / 256), 256, 0, stream>>>(
            accbuf, ssum, bias, outH, N, H, C);
    };

    // layer1: 66(pad 96) -> 8x128 concat; GEMM consumes bufH before finalize rewrites it
    run_layer(bufH, FINP, w1t, as1, ad1, b1, 8, 128, bufA, bufB, bufH);
    // layer2: 1024 -> 128
    run_layer(bufH, 1024, w2t, as2, ad2, b2, 1, 128, bufA, bufB, bufH);
    // layer3: 128 -> 64
    run_layer(bufH, 128,  w3t, as3, ad3, b3, 1, 64,  bufA, bufB, bufH);

    risk_head<<<(N + 255) / 256, 256, 0, stream>>>(bufH, Wr, br, (float*)d_out, N, 64);
}
